// Encoder_26688926777651
// MI455X (gfx1250) — matrix operand order repaired and checked by reading, not compiled, not run on hardware
//
#include <hip/hip_runtime.h>
#include <hip/hip_bf16.h>

// ---------------------------------------------------------------------------
// Bidirectional 2-layer GRU encoder for MI455X (gfx1250, wave32, WMMA).
//
// Pipeline (all on `stream`, dependencies via stream order):
//   1) swizzle_bT   : Whh{0,1}[dir], Wih1[dir] f32 -> bf16 B-fragment tiles
//   2) gather_xp0   : one-hot einsum == gather of Wih0 columns (+bih0)
//   3) init_hstate  : zero global h double-buffer + sync counters
//   4) gru_scan L0  : 8 WGs x 2 dirs; per-step 64x3072x1024 GEMM split by
//                     column slice; h staged to LDS with async-to-LDS loads,
//                     exchanged between WGs via L2 + atomic barrier
//   5) xp1_gemm     : (16384x2048)x(2048x3072) WMMA GEMM -> xp for layer 1
//   6) init_hstate + gru_scan L1 -> final outputs
//
// Workspace (~588 MB): xpF | xpB | y0 | whhT bf16 | wih1T bf16 | hstate | cnt
// ---------------------------------------------------------------------------

typedef __attribute__((ext_vector_type(16))) __bf16 v16bf;
typedef __attribute__((ext_vector_type(8)))  __bf16 v8bf;
typedef __attribute__((ext_vector_type(8)))  float  v8f;

constexpr int BATCH = 64;
constexpr int TSEQ  = 256;
constexpr int HID   = 1024;
constexpr int GDIM  = 3072;   // 3*HID
constexpr int VOC   = 128;
constexpr int IN1   = 2048;   // 2*HID
constexpr int NWG   = 8;      // column-slice workgroups per direction

constexpr size_t XP_FLOATS       = (size_t)BATCH * TSEQ * GDIM;
constexpr size_t Y0_FLOATS       = (size_t)BATCH * TSEQ * 2 * HID;
constexpr size_t WHH_TILE_ELEMS  = (size_t)192 * 32 * 512;   // per dir-layer
constexpr size_t WIH1_TILE_ELEMS = (size_t)192 * 64 * 512;   // per dir
constexpr size_t HSTATE_ELEMS    = (size_t)2 * 2 * BATCH * HID; // dir x buf

constexpr int HP = 1032;                          // LDS pitch (bf16), no conflicts
constexpr int SCAN_LDS_BYTES = 64 * HP * 2;       // 132,096 B (one h image)

__device__ __forceinline__ v8f wmma_bf16(v16bf a, v16bf b, v8f c) {
  return __builtin_amdgcn_wmma_f32_16x16x32_bf16(false, a, false, b,
                                                 (short)0, c, false, false);
}

// Async global->LDS copy, 16B per lane (ASYNCcnt-tracked, no VGPR round trip).
// VDST = LDS byte address (low 32 bits of generic LDS pointer), VADDR = 64-bit
// global address.  CDNA5 ISA.
__device__ __forceinline__ void async_g2l_b128(void* ldsPtr, const void* gPtr) {
  uint32_t lds32 = (uint32_t)(uintptr_t)ldsPtr;
  asm volatile("global_load_async_to_lds_b128 %0, %1, off"
               :: "v"(lds32), "v"((uint64_t)(uintptr_t)gPtr)
               : "memory");
}
__device__ __forceinline__ void wait_async0() {
#if __has_builtin(__builtin_amdgcn_s_wait_asynccnt)
  __builtin_amdgcn_s_wait_asynccnt(0);
#else
  asm volatile("s_wait_asynccnt 0x0" ::: "memory");
#endif
}

// --- 1) weight swizzle: W (rows=GDIM, cols=K) -> bf16 B-fragment tiles ------
// Consumer (lane L) reads 16 contiguous bf16 at tile*512 + L*16 where
// n = n0 + (L&15), k = k0 + (L&16) + s  (contiguous k order, same as the A loads; 32x16 tile).
__global__ void swizzle_bT(const float* __restrict__ W, __bf16* __restrict__ out,
                           int K, int kTiles) {
  int tid  = threadIdx.x;
  int tile = blockIdx.x * 8 + (tid >> 5);
  int L    = tid & 31;
  int nt   = tile / kTiles;
  int kt   = tile - nt * kTiles;
  int n    = nt * 16 + (L & 15);
  int kb   = kt * 32 + (L & 16);
  const float* src = W + (size_t)n * K + kb;
  __bf16* dst = out + (size_t)tile * 512 + (size_t)L * 16;
#pragma unroll
  for (int s = 0; s < 16; ++s) dst[s] = (__bf16)src[s];
}

// --- 2) layer-0 input projection: one-hot einsum == column gather -----------
__global__ void gather_xp0(const int* __restrict__ x,
                           const float* __restrict__ Wih0,
                           const float* __restrict__ bih0,
                           float* __restrict__ xpF, float* __restrict__ xpB) {
  int bt  = blockIdx.x;
  int dir = blockIdx.y;
  int b   = bt >> 8, ts = bt & 255;
  int t_in = dir ? (TSEQ - 1 - ts) : ts;
  int tok  = x[b * TSEQ + t_in];
  const float* Wc = Wih0 + (size_t)dir * GDIM * VOC + tok;
  const float* bi = bih0 + dir * GDIM;
  float* out = (dir ? xpB : xpF) + (size_t)bt * GDIM;
  for (int g = threadIdx.x; g < GDIM; g += blockDim.x)
    __builtin_nontemporal_store(Wc[(size_t)g * VOC] + bi[g], &out[g]);
}

// --- 3) zero h state double-buffer and barrier counters ---------------------
__global__ void init_hstate(__bf16* __restrict__ hstate, uint32_t* __restrict__ cnt) {
  size_t idx = (size_t)blockIdx.x * blockDim.x + threadIdx.x;
  uint32_t* p = (uint32_t*)hstate;            // HSTATE_ELEMS/2 dwords
  if (idx < HSTATE_ELEMS / 2) p[idx] = 0u;
  if (idx < 2) cnt[idx] = 0u;
}

// --- 5) layer-1 input projection GEMM: xp = y0 @ Wih1^T + bih1 --------------
__global__ void __launch_bounds__(256) xp1_gemm(
    const float* __restrict__ y0, const __bf16* __restrict__ wT,
    const float* __restrict__ bih1,
    float* __restrict__ xpF, float* __restrict__ xpB) {
  __shared__ __bf16 As[64 * 40];                 // 64x32 A block, pitch 40
  const int tid = threadIdx.x, L = tid & 31, w = tid >> 5;
  const int mi = w & 3, nh = w >> 2;
  const int dir = blockIdx.z;
  const int bt0 = blockIdx.x * 64;
  const __bf16* wTd = wT + (size_t)dir * WIH1_TILE_ELEMS;

  v8f acc[4];
#pragma unroll
  for (int p = 0; p < 4; ++p) acc[p] = {};

  const int mrow = mi * 16 + (L & 15);
  const int koff = (L & 16) ? 16 : 0;

  for (int kt = 0; kt < 64; ++kt) {
    const int k0 = kt * 32;
#pragma unroll
    for (int e = 0; e < 8; ++e) {
      int idx = tid * 8 + e;
      int r = idx >> 5, c = idx & 31;
      As[r * 40 + c] = (__bf16)y0[(size_t)(bt0 + r) * (2 * HID) + k0 + c];
    }
    __syncthreads();
    union { v16bf v; v8bf h[2]; } A;
    const __bf16* ap = &As[mrow * 40 + koff];
    A.h[0] = *(const v8bf*)ap;
    A.h[1] = *(const v8bf*)(ap + 8);
#pragma unroll
    for (int p = 0; p < 4; ++p) {
      int nt = blockIdx.y * 8 + nh * 4 + p;
      v16bf Bf = *(const v16bf*)(wTd + ((size_t)(nt * 64 + kt)) * 512 + L * 16);
      acc[p] = wmma_bf16(A.v, Bf, acc[p]);
    }
    __syncthreads();
  }

  // epilogue: time reversal folded into one signed row stride, NT stores
  float* out = dir ? xpB : xpF;
  const int b0    = bt0 >> 8;
  const int tbase = bt0 & 255;                   // rows are t = tbase + m
  const int sbase = dir ? (TSEQ - 1 - tbase) : tbase;
  const ptrdiff_t ostep = dir ? -(ptrdiff_t)GDIM : (ptrdiff_t)GDIM;
  const int mbase = mi * 16 + ((L & 16) ? 8 : 0);
#pragma unroll
  for (int p = 0; p < 4; ++p) {
    int nt = blockIdx.y * 8 + nh * 4 + p;
    int j  = nt * 16 + (L & 15);
    float bias = bih1[dir * GDIM + j];
    float* ob = out + ((size_t)b0 * TSEQ + sbase) * GDIM + j
                    + (ptrdiff_t)mbase * ostep;
#pragma unroll
    for (int i = 0; i < 8; ++i)
      __builtin_nontemporal_store(acc[p][i] + bias, ob + (ptrdiff_t)i * ostep);
  }
}

// --- 4/6) GRU scan: NWG column-slice WGs per direction, L2-barrier synced ---
// WG cw owns h columns [cw*128, cw*128+128) (matching r/z/n groups -> gating
// is WG-local). Per step: async-stage full h (global dbl-buf, L2-hot) into
// LDS, 64x384x1024 GEMM slice via WMMA, fused gating, write h slice to the
// other global buffer, then device-scope fence + per-direction atomic barrier.
__global__ void __launch_bounds__(512) gru_scan(
    const float* __restrict__ xpF, const float* __restrict__ xpB,
    const __bf16* __restrict__ whhT, const float* __restrict__ bhh,
    float* __restrict__ y, float* __restrict__ hOut,
    __bf16* __restrict__ hstate, uint32_t* __restrict__ cnt) {
  extern __shared__ __align__(16) char smem[];
  __bf16* hlds = (__bf16*)smem;

  const int cw  = blockIdx.x;                    // 0..NWG-1 column slice
  const int dir = blockIdx.y;
  const int tid = threadIdx.x, L = tid & 31, w = tid >> 5;
  const int mi = w & 3, cg = w >> 2;             // 4 M-subtiles x 4 col groups
  const int m0 = mi * 16;
  const __bf16* wTd  = whhT + (size_t)dir * WHH_TILE_ELEMS;
  const float*  xp   = dir ? xpB : xpF;
  const float*  bhhD = bhh + dir * GDIM;
  __bf16* hb = hstate + (size_t)dir * (2 * BATCH * HID);
  uint32_t* cd = cnt + dir;

  const int mrow = m0 + (L & 15);
  const int koff = (L & 16) ? 16 : 0;
  const int t0 = dir ? (TSEQ - 1) : 0;
  const ptrdiff_t ystep = dir ? -(ptrdiff_t)(2 * HID) : (ptrdiff_t)(2 * HID);

  // precompute per-(q,i) state so the 256-step loop is pure strength-reduction
  float brq[2], bzq[2], bnq[2];
  const float* xptr[2][8];
  float*       yptr[2][8];
  int gho[2][8];   // global h offset b*HID + j
  int lho[2][8];   // LDS    h offset b*HP  + j
#pragma unroll
  for (int q = 0; q < 2; ++q) {
    const int nb = cw * 8 + cg * 2 + q;
    const int j  = nb * 16 + (L & 15);
    brq[q] = bhhD[j]; bzq[q] = bhhD[j + HID]; bnq[q] = bhhD[j + 2 * HID];
#pragma unroll
    for (int i = 0; i < 8; ++i) {
      const int b = m0 + i + ((L & 16) ? 8 : 0);
      xptr[q][i] = xp + (size_t)b * TSEQ * GDIM + j;
      yptr[q][i] = y + ((size_t)b * TSEQ + t0) * (2 * HID) + dir * HID + j;
      gho[q][i]  = b * HID + j;
      lho[q][i]  = b * HP + j;
    }
  }

  uint32_t target = NWG;
  size_t tsoff = 0;
  int cur = 0;

  for (int ts = 0; ts < TSEQ; ++ts) {
    const __bf16* hc = hb + cur * (BATCH * HID);
    __bf16*       hn = hb + (1 - cur) * (BATCH * HID);

    // async-stage full h image into LDS (64 x 1024 bf16, pitch HP)
#pragma unroll 4
    for (int idx = tid; idx < 64 * 128; idx += 512) {
      int r = idx >> 7, cc = idx & 127;
      async_g2l_b128(hlds + r * HP + cc * 8, hc + r * HID + cc * 8);
    }
    wait_async0();          // own lanes' async loads landed in LDS
    __syncthreads();        // everyone's loads landed

#pragma unroll
    for (int q = 0; q < 2; ++q) {
      const int nb = cw * 8 + cg * 2 + q;
      v8f ar = {}, az = {}, an = {};
      const __bf16* arow  = hlds + (size_t)mrow * HP + koff;
      const __bf16* bbase = wTd + (size_t)nb * 32 * 512 + (size_t)L * 16;
#pragma unroll 2
      for (int kt = 0; kt < 32; ++kt) {
        union { v16bf v; v8bf h[2]; } A;
        const __bf16* ap = arow + kt * 32;
        A.h[0] = *(const v8bf*)ap;               // ds_load_b128
        A.h[1] = *(const v8bf*)(ap + 8);         // ds_load_b128
        const __bf16* bp = bbase + (size_t)kt * 512;
        v16bf Br = *(const v16bf*)bp;                              // r cols
        v16bf Bz = *(const v16bf*)(bp + (size_t)64  * 32 * 512);   // z cols
        v16bf Bn = *(const v16bf*)(bp + (size_t)128 * 32 * 512);   // n cols
        ar = wmma_bf16(A.v, Br, ar);
        az = wmma_bf16(A.v, Bz, az);
        an = wmma_bf16(A.v, Bn, an);
      }
#pragma unroll
      for (int i = 0; i < 8; ++i) {
        const float* xr_p = xptr[q][i] + tsoff;
        float xr = __builtin_nontemporal_load(xr_p);
        float xz = __builtin_nontemporal_load(xr_p + HID);
        float xn = __builtin_nontemporal_load(xr_p + 2 * HID);
        if (ts + 1 < TSEQ)                        // global_prefetch_b8
          __builtin_prefetch(xr_p + GDIM, 0, 1);
        float rg = 1.0f / (1.0f + __expf(-(xr + ar[i] + brq[q])));
        float zg = 1.0f / (1.0f + __expf(-(xz + az[i] + bzq[q])));
        float ng = tanhf(xn + rg * (an[i] + bnq[q]));
        float hold = (float)hlds[lho[q][i]];
        float hnew = (1.0f - zg) * ng + zg * hold;
        hn[gho[q][i]] = (__bf16)hnew;             // next state (global slice)
        yptr[q][i][0] = hnew;
        if (ts == TSEQ - 1)
          hOut[(size_t)dir * (BATCH * HID) + gho[q][i]] = hnew;
        yptr[q][i] += ystep;
      }
    }

    // inter-WG barrier for this direction (release h writes, acquire h reads)
    __threadfence();
    __syncthreads();
    if (tid == 0) {
      __hip_atomic_fetch_add(cd, 1u, __ATOMIC_RELEASE, __HIP_MEMORY_SCOPE_AGENT);
      while (__hip_atomic_load(cd, __ATOMIC_ACQUIRE, __HIP_MEMORY_SCOPE_AGENT) < target)
        __builtin_amdgcn_s_sleep(2);
    }
    __syncthreads();
    __threadfence();

    target += NWG;
    tsoff  += GDIM;
    cur ^= 1;
  }
}

// ---------------------------------------------------------------------------
extern "C" void kernel_launch(void* const* d_in, const int* in_sizes, int n_in,
                              void* d_out, int out_size, void* d_ws, size_t ws_size,
                              hipStream_t stream) {
  (void)in_sizes; (void)n_in; (void)out_size; (void)ws_size;

  const int*   x    = (const int*)  d_in[0];
  const float* Wih0 = (const float*)d_in[1];
  const float* Whh0 = (const float*)d_in[2];
  const float* bih0 = (const float*)d_in[3];
  const float* bhh0 = (const float*)d_in[4];
  const float* Wih1 = (const float*)d_in[5];
  const float* Whh1 = (const float*)d_in[6];
  const float* bih1 = (const float*)d_in[7];
  const float* bhh1 = (const float*)d_in[8];

  float*  xpF    = (float*)d_ws;
  float*  xpB    = xpF + XP_FLOATS;
  float*  y0     = xpB + XP_FLOATS;
  __bf16* whhT   = (__bf16*)(y0 + Y0_FLOATS);
  __bf16* wih1T  = whhT + 4 * WHH_TILE_ELEMS;
  __bf16* hstate = wih1T + 2 * WIH1_TILE_ELEMS;
  uint32_t* cnt  = (uint32_t*)(hstate + HSTATE_ELEMS);

  float* outSeq = (float*)d_out;               // (B, T, 2H)
  float* outHid = outSeq + Y0_FLOATS;          // (4, B, H): [hf0, hb0, hf1, hb1]

  // 1) weight swizzle to bf16 fragment tiles
  swizzle_bT<<<768, 256, 0, stream>>>(Whh0,                       whhT + 0 * WHH_TILE_ELEMS, HID, 32);
  swizzle_bT<<<768, 256, 0, stream>>>(Whh0 + (size_t)GDIM * HID,  whhT + 1 * WHH_TILE_ELEMS, HID, 32);
  swizzle_bT<<<768, 256, 0, stream>>>(Whh1,                       whhT + 2 * WHH_TILE_ELEMS, HID, 32);
  swizzle_bT<<<768, 256, 0, stream>>>(Whh1 + (size_t)GDIM * HID,  whhT + 3 * WHH_TILE_ELEMS, HID, 32);
  swizzle_bT<<<1536, 256, 0, stream>>>(Wih1,                      wih1T,                     IN1, 64);
  swizzle_bT<<<1536, 256, 0, stream>>>(Wih1 + (size_t)GDIM * IN1, wih1T + WIH1_TILE_ELEMS,   IN1, 64);

  // 2) layer-0 input projection (gather)
  gather_xp0<<<dim3(BATCH * TSEQ, 2), 256, 0, stream>>>(x, Wih0, bih0, xpF, xpB);

  hipFuncSetAttribute(reinterpret_cast<const void*>(gru_scan),
                      hipFuncAttributeMaxDynamicSharedMemorySize, SCAN_LDS_BYTES);

  // 3+4) layer-0 scan
  init_hstate<<<(HSTATE_ELEMS / 2 + 255) / 256, 256, 0, stream>>>(hstate, cnt);
  gru_scan<<<dim3(NWG, 2), 512, SCAN_LDS_BYTES, stream>>>(
      xpF, xpB, whhT, bhh0, y0, outHid, hstate, cnt);

  // 5) layer-1 input projection GEMM
  xp1_gemm<<<dim3(256, 24, 2), 256, 0, stream>>>(y0, wih1T, bih1, xpF, xpB);

  // 6) layer-1 scan -> final outputs
  init_hstate<<<(HSTATE_ELEMS / 2 + 255) / 256, 256, 0, stream>>>(hstate, cnt);
  gru_scan<<<dim3(NWG, 2), 512, SCAN_LDS_BYTES, stream>>>(
      xpF, xpB, whhT + 2 * WHH_TILE_ELEMS, bhh1, outSeq,
      outHid + 2 * (size_t)BATCH * HID, hstate, cnt);
}
